// Decoder_22849226015156
// MI455X (gfx1250) — compile-verified, matrix operand order checked by reading, not run on hardware
//
#include <hip/hip_runtime.h>
#include <math.h>

// ---------------- problem constants ----------------
#define H_DIM   1024
#define D_DIM   256
#define T_STEPS 64
#define B_DIM   256
#define FC1_DIM 1024
#define FC2_DIM 512
#define LN_EPS  1e-5f

// ---------------- WMMA types ----------------
typedef __attribute__((ext_vector_type(16))) __bf16 bf16x16;
typedef __attribute__((ext_vector_type(8)))  float  v8f;

union FragBF {
    bf16x16 v;
    uint4   q[2];
};

// fp32 -> bf16 (round to nearest even)
__device__ __forceinline__ unsigned short f2bf(float f) {
    union { float f; unsigned u; } x;
    x.f = f;
    unsigned r = x.u + 0x7FFFu + ((x.u >> 16) & 1u);
    return (unsigned short)(r >> 16);
}

__device__ __forceinline__ float sigmoid_f(float x) {
    return 1.0f / (1.0f + expf(-x));
}

// =====================================================================
// Weight conversion: fp32 -> bf16
// =====================================================================
__global__ void cvt_bf16_kernel(const float* __restrict__ src,
                                unsigned short* __restrict__ dst, int n) {
    int i = blockIdx.x * blockDim.x + threadIdx.x;
    if (i < n) dst[i] = f2bf(src[i]);
}

// =====================================================================
// State init: h0/h1 from `hidden`, x (fed-back input) = 0
// =====================================================================
__global__ void init_state_kernel(const float* __restrict__ hidden,
                                  float* __restrict__ h0, float* __restrict__ h1,
                                  unsigned short* __restrict__ h0bf,
                                  unsigned short* __restrict__ h1bf,
                                  unsigned short* __restrict__ xbf) {
    int i = blockIdx.x * blockDim.x + threadIdx.x;
    const int NH = B_DIM * H_DIM;
    if (i < NH) {
        float a = hidden[i];
        float b = hidden[NH + i];
        h0[i] = a;  h0bf[i] = f2bf(a);
        h1[i] = b;  h1bf[i] = f2bf(b);
    }
    if (i < B_DIM * D_DIM) xbf[i] = 0;  // bf16 zero
}

// =====================================================================
// bf16 WMMA GEMM:  C[M,N] = A[M,K] * W[N,K]^T + bias[N]
//   workgroup: 256 threads (8 wave32), tile 128(M) x 64(N), K-step 32
//   each wave computes a 32x32 patch via 2x2 v_wmma_f32_16x16x32_bf16
// =====================================================================
#define GT_M   128
#define GT_N   64
#define GT_K   32
#define LDS_K  40   // padded row stride (bf16 elems): 80B, 16B-aligned

__global__ __launch_bounds__(256) void gemm_bf16_wmma(
    const unsigned short* __restrict__ A,   // [M,K] bf16
    const unsigned short* __restrict__ W,   // [N,K] bf16
    const float* __restrict__ bias,         // [N]
    float* __restrict__ C,                  // [M,N]
    int M, int N, int K)
{
    __shared__ unsigned short Xs[GT_M * LDS_K];
    __shared__ unsigned short Ws[GT_N * LDS_K];

    const int tid   = threadIdx.x;
    const int lane  = tid & 31;
    const int wave  = tid >> 5;
    const int waveM = wave & 3;   // 4 waves along M
    const int waveN = wave >> 2;  // 2 waves along N

    const int mBase = blockIdx.y * GT_M;
    const int nBase = blockIdx.x * GT_N;

    v8f zero = {};
    v8f acc[2][2];
    acc[0][0] = zero; acc[0][1] = zero; acc[1][0] = zero; acc[1][1] = zero;

    const int lrow  = lane & 15;
    const int khalf = (lane >> 4) * 8;   // 0 or 8 (bf16 elems)

    for (int k0 = 0; k0 < K; k0 += GT_K) {
        // ---- stage A tile (128x32 bf16 = 512 x 16B chunks, 2 per thread)
        #pragma unroll
        for (int c = 0; c < 2; ++c) {
            int chunk = tid + c * 256;
            int row = chunk >> 2;
            int kc  = chunk & 3;
            const uint4* g = (const uint4*)(A + (size_t)(mBase + row) * K + k0 + kc * 8);
            *(uint4*)(&Xs[row * LDS_K + kc * 8]) = *g;
        }
        // ---- stage W tile (64x32 bf16 = 256 x 16B chunks, 1 per thread)
        {
            int row = tid >> 2;
            int kc  = tid & 3;
            const uint4* g = (const uint4*)(W + (size_t)(nBase + row) * K + k0 + kc * 8);
            *(uint4*)(&Ws[row * LDS_K + kc * 8]) = *g;
        }
        __syncthreads();

        // ---- build fragments per ISA 16-bit A/B layout and issue WMMA
        FragBF a[2], b[2];
        #pragma unroll
        for (int i = 0; i < 2; ++i) {
            int r = waveM * 32 + i * 16 + lrow;
            a[i].q[0] = *(const uint4*)(&Xs[r * LDS_K + khalf]);
            a[i].q[1] = *(const uint4*)(&Xs[r * LDS_K + khalf + 16]);
        }
        #pragma unroll
        for (int j = 0; j < 2; ++j) {
            int r = waveN * 32 + j * 16 + lrow;
            b[j].q[0] = *(const uint4*)(&Ws[r * LDS_K + khalf]);
            b[j].q[1] = *(const uint4*)(&Ws[r * LDS_K + khalf + 16]);
        }
        #pragma unroll
        for (int i = 0; i < 2; ++i)
            #pragma unroll
            for (int j = 0; j < 2; ++j)
                acc[i][j] = __builtin_amdgcn_wmma_f32_16x16x32_bf16(
                    false, a[i].v, false, b[j].v,
                    (short)0, acc[i][j], false, false);
        __syncthreads();
    }

    // ---- epilogue: 16x16 f32 C layout (lane = col; vgpr r = row, +8 for hi lanes)
    const int rsel = (lane >> 4) * 8;
    #pragma unroll
    for (int j = 0; j < 2; ++j) {
        int n = nBase + waveN * 32 + j * 16 + lrow;
        float bv = bias[n];
        #pragma unroll
        for (int i = 0; i < 2; ++i) {
            int m0 = mBase + waveM * 32 + i * 16 + rsel;
            #pragma unroll
            for (int r = 0; r < 8; ++r)
                C[(size_t)(m0 + r) * N + n] = acc[i][j][r] + bv;
        }
    }
}

// =====================================================================
// GRU gate fusion:  h = (1-z)*n + z*h   (in-place), also emit bf16 h
// =====================================================================
__global__ void gru_gate_kernel(const float* __restrict__ gi,
                                const float* __restrict__ gh,
                                float* __restrict__ h,
                                unsigned short* __restrict__ hbf) {
    int idx = blockIdx.x * blockDim.x + threadIdx.x;   // over B*H
    if (idx >= B_DIM * H_DIM) return;
    int b = idx >> 10;           // / H_DIM
    int j = idx & (H_DIM - 1);
    const float* gib = gi + (size_t)b * 3 * H_DIM;
    const float* ghb = gh + (size_t)b * 3 * H_DIM;
    float r = sigmoid_f(gib[j] + ghb[j]);
    float z = sigmoid_f(gib[H_DIM + j] + ghb[H_DIM + j]);
    float n = tanhf(gib[2 * H_DIM + j] + r * ghb[2 * H_DIM + j]);
    float hv = h[idx];
    float out = (1.0f - z) * n + z * hv;
    h[idx]   = out;
    hbf[idx] = f2bf(out);
}

// =====================================================================
// LayerNorm + exact GELU, one row per workgroup (256 threads, wave32)
// =====================================================================
__global__ __launch_bounds__(256) void ln_gelu_kernel(
    const float* __restrict__ x, const float* __restrict__ gamma,
    const float* __restrict__ beta, unsigned short* __restrict__ ybf, int N)
{
    __shared__ float redS[8];
    __shared__ float redS2[8];
    __shared__ float stats[2];

    const int b   = blockIdx.x;
    const int tid = threadIdx.x;
    const float* xb = x + (size_t)b * N;

    float s = 0.0f, s2 = 0.0f;
    for (int i = tid; i < N; i += 256) {
        float v = xb[i];
        s += v; s2 += v * v;
    }
    #pragma unroll
    for (int o = 16; o > 0; o >>= 1) {
        s  += __shfl_down(s,  o, 32);
        s2 += __shfl_down(s2, o, 32);
    }
    if ((tid & 31) == 0) { redS[tid >> 5] = s; redS2[tid >> 5] = s2; }
    __syncthreads();
    if (tid == 0) {
        float a = 0.0f, c = 0.0f;
        #pragma unroll
        for (int i = 0; i < 8; ++i) { a += redS[i]; c += redS2[i]; }
        float mu  = a / (float)N;
        float var = c / (float)N - mu * mu;
        stats[0] = mu;
        stats[1] = rsqrtf(var + LN_EPS);
    }
    __syncthreads();
    const float mu  = stats[0];
    const float inv = stats[1];
    for (int i = tid; i < N; i += 256) {
        float v = (xb[i] - mu) * inv * gamma[i] + beta[i];
        float g = 0.5f * v * (1.0f + erff(v * 0.70710678118654752f));  // exact GELU
        ybf[(size_t)b * N + i] = f2bf(g);
    }
}

// =====================================================================
// Softmax over D=256, store to out[b,t,:] and bf16 feedback input
// =====================================================================
__global__ __launch_bounds__(256) void softmax_store_kernel(
    const float* __restrict__ logits, float* __restrict__ out,
    unsigned short* __restrict__ xbf, int t)
{
    __shared__ float red[8];
    __shared__ float stat;

    const int b   = blockIdx.x;
    const int tid = threadIdx.x;
    float v = logits[(size_t)b * D_DIM + tid];

    float m = v;
    #pragma unroll
    for (int o = 16; o > 0; o >>= 1) m = fmaxf(m, __shfl_down(m, o, 32));
    if ((tid & 31) == 0) red[tid >> 5] = m;
    __syncthreads();
    if (tid == 0) {
        float mm = red[0];
        #pragma unroll
        for (int i = 1; i < 8; ++i) mm = fmaxf(mm, red[i]);
        stat = mm;
    }
    __syncthreads();
    float e = expf(v - stat);
    float s = e;
    #pragma unroll
    for (int o = 16; o > 0; o >>= 1) s += __shfl_down(s, o, 32);
    if ((tid & 31) == 0) red[tid >> 5] = s;
    __syncthreads();
    if (tid == 0) {
        float ss = 0.0f;
        #pragma unroll
        for (int i = 0; i < 8; ++i) ss += red[i];
        stat = ss;
    }
    __syncthreads();
    float p = e / stat;
    out[((size_t)b * T_STEPS + t) * D_DIM + tid] = p;
    xbf[(size_t)b * D_DIM + tid] = f2bf(p);
}

// =====================================================================
// Host orchestration
// =====================================================================
extern "C" void kernel_launch(void* const* d_in, const int* in_sizes, int n_in,
                              void* d_out, int out_size, void* d_ws, size_t ws_size,
                              hipStream_t stream) {
    (void)in_sizes; (void)n_in; (void)out_size; (void)ws_size;

    const float* hidden = (const float*)d_in[0];
    const float* W_ih0  = (const float*)d_in[1];
    const float* W_hh0  = (const float*)d_in[2];
    const float* b_ih0  = (const float*)d_in[3];
    const float* b_hh0  = (const float*)d_in[4];
    const float* W_ih1  = (const float*)d_in[5];
    const float* W_hh1  = (const float*)d_in[6];
    const float* b_ih1  = (const float*)d_in[7];
    const float* b_hh1  = (const float*)d_in[8];
    const float* fc1_w  = (const float*)d_in[9];
    const float* fc1_b  = (const float*)d_in[10];
    const float* ln1_g  = (const float*)d_in[11];
    const float* ln1_b  = (const float*)d_in[12];
    const float* fc2_w  = (const float*)d_in[13];
    const float* fc2_b  = (const float*)d_in[14];
    const float* ln2_g  = (const float*)d_in[15];
    const float* ln2_b  = (const float*)d_in[16];
    const float* fc3_w  = (const float*)d_in[17];
    const float* fc3_b  = (const float*)d_in[18];
    float* out = (float*)d_out;

    // ---- workspace carve-up
    char*  ws  = (char*)d_ws;
    size_t off = 0;
    auto alloc = [&](size_t bytes) -> void* {
        void* p = ws + off;
        off += (bytes + 255) & ~(size_t)255;
        return p;
    };
    const int G3H = 3 * H_DIM;  // 3072

    unsigned short* Wih0bf = (unsigned short*)alloc((size_t)G3H * D_DIM * 2);
    unsigned short* Whh0bf = (unsigned short*)alloc((size_t)G3H * H_DIM * 2);
    unsigned short* Wih1bf = (unsigned short*)alloc((size_t)G3H * H_DIM * 2);
    unsigned short* Whh1bf = (unsigned short*)alloc((size_t)G3H * H_DIM * 2);
    unsigned short* fc1wbf = (unsigned short*)alloc((size_t)FC1_DIM * H_DIM * 2);
    unsigned short* fc2wbf = (unsigned short*)alloc((size_t)FC2_DIM * FC1_DIM * 2);
    unsigned short* fc3wbf = (unsigned short*)alloc((size_t)D_DIM * FC2_DIM * 2);

    float*          h0     = (float*)alloc((size_t)B_DIM * H_DIM * 4);
    float*          h1     = (float*)alloc((size_t)B_DIM * H_DIM * 4);
    unsigned short* h0bf   = (unsigned short*)alloc((size_t)B_DIM * H_DIM * 2);
    unsigned short* h1bf   = (unsigned short*)alloc((size_t)B_DIM * H_DIM * 2);
    unsigned short* xbf    = (unsigned short*)alloc((size_t)B_DIM * D_DIM * 2);
    float*          gi     = (float*)alloc((size_t)B_DIM * G3H * 4);
    float*          gh     = (float*)alloc((size_t)B_DIM * G3H * 4);
    float*          a1     = (float*)alloc((size_t)B_DIM * FC1_DIM * 4);
    unsigned short* a1bf   = (unsigned short*)alloc((size_t)B_DIM * FC1_DIM * 2);
    float*          a2     = (float*)alloc((size_t)B_DIM * FC2_DIM * 4);
    unsigned short* a2bf   = (unsigned short*)alloc((size_t)B_DIM * FC2_DIM * 2);
    float*          logits = (float*)alloc((size_t)B_DIM * D_DIM * 4);

    // ---- one-time (per call) weight conversion fp32 -> bf16
    auto cvt = [&](const float* src, unsigned short* dst, int n) {
        cvt_bf16_kernel<<<(n + 255) / 256, 256, 0, stream>>>(src, dst, n);
    };
    cvt(W_ih0, Wih0bf, G3H * D_DIM);
    cvt(W_hh0, Whh0bf, G3H * H_DIM);
    cvt(W_ih1, Wih1bf, G3H * H_DIM);
    cvt(W_hh1, Whh1bf, G3H * H_DIM);
    cvt(fc1_w, fc1wbf, FC1_DIM * H_DIM);
    cvt(fc2_w, fc2wbf, FC2_DIM * FC1_DIM);
    cvt(fc3_w, fc3wbf, D_DIM * FC2_DIM);

    init_state_kernel<<<(B_DIM * H_DIM + 255) / 256, 256, 0, stream>>>(
        hidden, h0, h1, h0bf, h1bf, xbf);

    auto gemm = [&](const unsigned short* A, const unsigned short* W,
                    const float* bias, float* C, int M, int N, int K) {
        dim3 grid(N / GT_N, M / GT_M);
        gemm_bf16_wmma<<<grid, 256, 0, stream>>>(A, W, bias, C, M, N, K);
    };

    const int nBH = B_DIM * H_DIM;

    for (int t = 0; t < T_STEPS; ++t) {
        // GRU layer 0: x[B,256] ; h0[B,1024]
        gemm(xbf,  Wih0bf, b_ih0, gi, B_DIM, G3H, D_DIM);
        gemm(h0bf, Whh0bf, b_hh0, gh, B_DIM, G3H, H_DIM);
        gru_gate_kernel<<<(nBH + 255) / 256, 256, 0, stream>>>(gi, gh, h0, h0bf);

        // GRU layer 1: h0[B,1024] ; h1[B,1024]
        gemm(h0bf, Wih1bf, b_ih1, gi, B_DIM, G3H, H_DIM);
        gemm(h1bf, Whh1bf, b_hh1, gh, B_DIM, G3H, H_DIM);
        gru_gate_kernel<<<(nBH + 255) / 256, 256, 0, stream>>>(gi, gh, h1, h1bf);

        // FC head
        gemm(h1bf, fc1wbf, fc1_b, a1, B_DIM, FC1_DIM, H_DIM);
        ln_gelu_kernel<<<B_DIM, 256, 0, stream>>>(a1, ln1_g, ln1_b, a1bf, FC1_DIM);
        gemm(a1bf, fc2wbf, fc2_b, a2, B_DIM, FC2_DIM, FC1_DIM);
        ln_gelu_kernel<<<B_DIM, 256, 0, stream>>>(a2, ln2_g, ln2_b, a2bf, FC2_DIM);
        gemm(a2bf, fc3wbf, fc3_b, logits, B_DIM, D_DIM, FC2_DIM);

        // softmax -> out[:, t, :] and feedback input
        softmax_store_kernel<<<B_DIM, 256, 0, stream>>>(logits, out, xbf, t);
    }
}
